// PointerHead_20332375179520
// MI455X (gfx1250) — matrix operand order repaired and checked by reading, not compiled, not run on hardware
//
#include <hip/hip_runtime.h>
#include <hip/hip_bf16.h>

typedef __attribute__((ext_vector_type(16))) _Float16 v16h;
typedef __attribute__((ext_vector_type(8)))  _Float16 v8h;
typedef __attribute__((ext_vector_type(8)))  float    v8f;

#define BB 8
#define NN 2048
#define MM 256

__device__ __forceinline__ float sigmoidf_(float x) { return 1.0f / (1.0f + __expf(-x)); }

// order-preserving float <-> uint encode (for combined max+argmax reduction)
__device__ __forceinline__ unsigned f2ord(float f) {
    unsigned u = __float_as_uint(f);
    return (u & 0x80000000u) ? ~u : (u | 0x80000000u);
}
__device__ __forceinline__ float ord2f(unsigned e) {
    unsigned u = (e & 0x80000000u) ? (e ^ 0x80000000u) : ~e;
    return __uint_as_float(u);
}

// Multi-tile WMMA matvec: NT row-tiles (rows row0 + t*16 .. +16) of mtx[rows x 256]
// times vec (f16 in LDS, only column 0 of the B operand is populated).
// One B build per k-chunk is shared by all NT tiles.
// Output: lanes with (lane&15)==0 hold 8 f32 per tile; lane0 -> rows +0..7,
// lane16 -> rows +8..15 (documented 16x16 f32 C/D layout).
template <int NT>
__device__ __forceinline__ void wmma_matvec_multi(const _Float16* __restrict__ mtx,
                                                  int row0, int ldm,
                                                  const _Float16* __restrict__ vec_h,
                                                  int lane, v8f* acc) {
    const int koA = (lane >> 4) * 8;    // A layout: lanes 0-15 K base 0, lanes 16-31 base 8
    const int koB = (lane >> 4) * 8;    // B layout: same K order as A (base 0 / 8, plus 16)
    const bool bld = (lane & 15) == 0;  // only column 0 of B carries the vector
    const _Float16* rowp[NT];
#pragma unroll
    for (int t = 0; t < NT; ++t)
        rowp[t] = mtx + (size_t)(row0 + t * 16 + (lane & 15)) * ldm;
#pragma unroll
    for (int t = 0; t < NT; ++t) acc[t] = (v8f){};
    for (int k0 = 0; k0 < MM; k0 += 32) {
        v16h b;
#pragma unroll
        for (int i = 0; i < 16; ++i) b[i] = (_Float16)0.f;
        if (bld) {  // two aligned 16B LDS loads, no conversion
            v8h b0 = *(const v8h*)(vec_h + k0 + koB);
            v8h b1 = *(const v8h*)(vec_h + k0 + 16 + koB);
#pragma unroll
            for (int i = 0; i < 8; ++i) { b[i] = b0[i]; b[8 + i] = b1[i]; }
        }
#pragma unroll
        for (int t = 0; t < NT; ++t) {
            v16h a;
            v8h lo = *(const v8h*)(rowp[t] + k0 + koA);
            v8h hi = *(const v8h*)(rowp[t] + k0 + 16 + koA);
#pragma unroll
            for (int i = 0; i < 8; ++i) { a[i] = lo[i]; a[8 + i] = hi[i]; }
            acc[t] = __builtin_amdgcn_wmma_f32_16x16x32_f16(false, a, false, b,
                                                            (short)0, acc[t], false, false);
        }
    }
}

// ---------- prep kernels ----------
__global__ void cvt_f16_kernel(const float* __restrict__ src, _Float16* __restrict__ dst, int n) {
    int i = blockIdx.x * blockDim.x + threadIdx.x;
    if (i < n) dst[i] = (_Float16)src[i];
}

// xph[j] = b_ih[j] + b_hh[j] + dot(decoder_init, w_ih[j,:])  (constant LSTM input proj)
__global__ void xph_kernel(const float* __restrict__ dec_init, const float* __restrict__ w_ih,
                           const float* __restrict__ b_ih, const float* __restrict__ b_hh,
                           float* __restrict__ xph) {
    int j = blockIdx.x * blockDim.x + threadIdx.x;
    if (j < 4 * MM) {
        float s = b_ih[j] + b_hh[j];
        const float* row = w_ih + (size_t)j * MM;
        for (int k = 0; k < MM; ++k) s += dec_init[k] * row[k];
        xph[j] = s;
    }
}

// keys[b][n][m] = sum_k ne[b][n][k] * Wk[m][k] + bk[m], stored f16 row-major (n-major).
// Dense WMMA: grid = B * (N/16) blocks, 256 threads = 8 waves, each wave 2 m-tiles.
__global__ __launch_bounds__(256) void keys_wmma_kernel(const float* __restrict__ ne,
                                                        const float* __restrict__ Wk,
                                                        const float* __restrict__ bk,
                                                        _Float16* __restrict__ keys) {
    const int lane = threadIdx.x & 31;
    const int wv   = threadIdx.x >> 5;
    const int b    = blockIdx.x >> 7;          // 128 n-tiles per batch
    const int n0   = (blockIdx.x & 127) * 16;
    const float* neb = ne + (size_t)b * NN * MM;
    const int koA = (lane >> 4) * 8;

    for (int mt = wv * 2; mt < wv * 2 + 2; ++mt) {
        const int m0 = mt * 16;
        v8f acc = {};
        const float* arow = neb + (size_t)(n0 + (lane & 15)) * MM;
        const float* brow = Wk  + (size_t)(m0 + (lane & 15)) * MM;  // B[k][m]=Wk[m][k]
        for (int k0 = 0; k0 < MM; k0 += 32) {
            v16h a, bv;
#pragma unroll
            for (int i = 0; i < 8; ++i) {
                a[i]     = (_Float16)arow[k0 + koA + i];
                a[8 + i] = (_Float16)arow[k0 + 16 + koA + i];
            }
#pragma unroll
            for (int i = 0; i < 8; ++i) {
                bv[i]     = (_Float16)brow[k0 + koA + i];
                bv[8 + i] = (_Float16)brow[k0 + 16 + koA + i];
            }
            acc = __builtin_amdgcn_wmma_f32_16x16x32_f16(false, a, false, bv,
                                                         (short)0, acc, false, false);
        }
        const float bkv = bk[m0 + (lane & 15)];
        const int m = m0 + (lane & 15);
        const int ro = (lane >> 4) * 8;
#pragma unroll
        for (int r = 0; r < 8; ++r) {
            int n = n0 + r + ro;
            keys[((size_t)b * NN + n) * MM + m] = (_Float16)(acc[r] + bkv);
        }
    }
}

// ---------- sequential decode: one workgroup per batch ----------
__global__ __launch_bounds__(1024) void decode_kernel(const _Float16* __restrict__ keys,
                                                      const _Float16* __restrict__ whh,
                                                      const _Float16* __restrict__ wqh,
                                                      const float* __restrict__ xph,
                                                      const float* __restrict__ z_g,
                                                      const float* __restrict__ hidden0,
                                                      const float* __restrict__ bq,
                                                      float* __restrict__ out) {
    __shared__ __align__(32) _Float16 h_h[MM];   // hidden state, f16 (B operand)
    __shared__ __align__(32) _Float16 q_h[MM];   // query, f16 (B operand)
    __shared__ float gates_s[4 * MM];
    __shared__ float scores_s[NN];
    __shared__ float used_s[NN];
    __shared__ unsigned long long red64[32];
    __shared__ float red32[32];

    const int tid  = threadIdx.x;
    const int lane = tid & 31;
    const int wv   = tid >> 5;
    const int b    = blockIdx.x;
    const _Float16* keysb = keys + (size_t)b * NN * MM;

    float c_reg = 0.f;  // LSTM cell state lives in a register of threads 0..255
    if (tid < MM) {
        h_h[tid] = (_Float16)hidden0[tid];
        c_reg = z_g[(size_t)b * MM + tid];
    }
    for (int n = tid; n < NN; n += 1024) used_s[n] = 0.f;
    __syncthreads();

    for (int t = 0; t < NN; ++t) {
        // ---- gates = xph + Whh @ h : 32 waves x 32 rows (2 tiles, shared B) ----
        {
            const int j0 = wv * 32;
            v8f acc[2];
            wmma_matvec_multi<2>(whh, j0, MM, h_h, lane, acc);
            if ((lane & 15) == 0) {
                const int ro = (lane >> 4) * 8;
#pragma unroll
                for (int r = 0; r < 8; ++r) {
                    gates_s[j0 + r + ro]      = acc[0][r] + xph[j0 + r + ro];
                    gates_s[j0 + 16 + r + ro] = acc[1][r] + xph[j0 + 16 + r + ro];
                }
            }
        }
        __syncthreads();
        // ---- LSTM cell update (i,f,g,o ordering) ----
        if (tid < MM) {
            float ig = sigmoidf_(gates_s[tid]);
            float fg = sigmoidf_(gates_s[MM + tid]);
            float gg = tanhf(gates_s[2 * MM + tid]);
            float og = sigmoidf_(gates_s[3 * MM + tid]);
            float cn = fg * c_reg + ig * gg;
            c_reg = cn;
            h_h[tid] = (_Float16)(og * tanhf(cn));
        }
        __syncthreads();
        // ---- q = Wq @ h + bq : waves 0..15, 16 rows each; stored f16 ----
        if (wv < 16) {
            const int j0 = wv * 16;
            v8f acc[1];
            wmma_matvec_multi<1>(wqh, j0, MM, h_h, lane, acc);
            if ((lane & 15) == 0) {
                const int ro = (lane >> 4) * 8;
#pragma unroll
                for (int r = 0; r < 8; ++r)
                    q_h[j0 + r + ro] = (_Float16)(acc[0][r] + bq[j0 + r + ro]);
            }
        }
        __syncthreads();
        // ---- scores = keys @ q : 128 tiles, 4 per wave sharing one B build ----
        {
            const int n0 = wv * 64;
            v8f acc[4];
            wmma_matvec_multi<4>(keysb, n0, MM, q_h, lane, acc);
            if ((lane & 15) == 0) {
                const int ro = (lane >> 4) * 8;
#pragma unroll
                for (int ti = 0; ti < 4; ++ti)
#pragma unroll
                    for (int r = 0; r < 8; ++r)
                        scores_s[n0 + ti * 16 + r + ro] = acc[ti][r];
            }
        }
        __syncthreads();
        // ---- masked softmax + argmax (first-index tie-break) ----
        float s0 = scores_s[tid]        - used_s[tid]        * 1e9f;
        float s1 = scores_s[tid + 1024] - used_s[tid + 1024] * 1e9f;
        unsigned long long k0 = ((unsigned long long)f2ord(s0) << 32) | (unsigned)(~tid);
        unsigned long long k1 = ((unsigned long long)f2ord(s1) << 32) | (unsigned)(~(tid + 1024));
        unsigned long long kk = (k0 > k1) ? k0 : k1;
        for (int off = 16; off; off >>= 1) {
            unsigned long long o = __shfl_down(kk, off, 32);
            if (o > kk) kk = o;
        }
        if (lane == 0) red64[wv] = kk;
        __syncthreads();
        if (tid < 32) {
            unsigned long long m = red64[tid];
            for (int off = 16; off; off >>= 1) {
                unsigned long long o = __shfl_down(m, off, 32);
                if (o > m) m = o;
            }
            if (tid == 0) red64[0] = m;
        }
        __syncthreads();
        const unsigned long long best = red64[0];
        const float maxv = ord2f((unsigned)(best >> 32));
        const int bestn = (int)(~(unsigned)(best & 0xFFFFFFFFu));
        float e0 = __expf(s0 - maxv), e1 = __expf(s1 - maxv);
        float ss = e0 + e1;
        for (int off = 16; off; off >>= 1) ss += __shfl_down(ss, off, 32);
        if (lane == 0) red32[wv] = ss;
        __syncthreads();
        if (tid < 32) {
            float m = red32[tid];
            for (int off = 16; off; off >>= 1) m += __shfl_down(m, off, 32);
            if (tid == 0) red32[0] = m;
        }
        __syncthreads();
        const float inv = 1.0f / red32[0];
        float* orow = out + ((size_t)b * NN + t) * NN;
        orow[tid] = e0 * inv;
        orow[tid + 1024] = e1 * inv;
        if (tid == 0) used_s[bestn] = 1.f;
        __syncthreads();
    }
}

extern "C" void kernel_launch(void* const* d_in, const int* in_sizes, int n_in,
                              void* d_out, int out_size, void* d_ws, size_t ws_size,
                              hipStream_t stream) {
    const float* node_emb = (const float*)d_in[0];
    const float* z_g      = (const float*)d_in[1];
    const float* dec_init = (const float*)d_in[2];
    const float* hidden0  = (const float*)d_in[3];
    const float* w_ih     = (const float*)d_in[4];
    const float* w_hh     = (const float*)d_in[5];
    const float* b_ih     = (const float*)d_in[6];
    const float* b_hh     = (const float*)d_in[7];
    const float* Wq       = (const float*)d_in[8];
    const float* bq       = (const float*)d_in[9];
    const float* Wk       = (const float*)d_in[10];
    const float* bk       = (const float*)d_in[11];

    char* ws = (char*)d_ws;
    _Float16* keys = (_Float16*)ws;                                        // 8 MB
    _Float16* whh  = (_Float16*)(ws + ((size_t)8 << 20));                  // 512 KB
    _Float16* wqh  = (_Float16*)(ws + ((size_t)8 << 20) + (512 << 10));    // 128 KB
    float*    xph  = (float*)   (ws + ((size_t)8 << 20) + (512 << 10) + (128 << 10)); // 4 KB

    cvt_f16_kernel<<<(4 * MM * MM + 255) / 256, 256, 0, stream>>>(w_hh, whh, 4 * MM * MM);
    cvt_f16_kernel<<<(MM * MM + 255) / 256, 256, 0, stream>>>(Wq, wqh, MM * MM);
    xph_kernel<<<4, 256, 0, stream>>>(dec_init, w_ih, b_ih, b_hh, xph);
    keys_wmma_kernel<<<BB * (NN / 16), 256, 0, stream>>>(node_emb, Wk, bk, keys);
    decode_kernel<<<BB, 1024, 0, stream>>>(keys, whh, wqh, xph, z_g, hidden0, bq,
                                           (float*)d_out);
}
